// GAT_2L_1056561955273
// MI455X (gfx1250) — matrix operand order repaired and checked by reading, not compiled, not run on hardware
//
#include <hip/hip_runtime.h>
#include <cstdint>
#include <cstddef>

// ---------------------------------------------------------------------------
// GATv2 2-layer network for MI455X (gfx1250, wave32, WMMA bf16)
// ---------------------------------------------------------------------------
#define GN 50000
#define GE 800000
#define GDIN 512
#define GPROJ 256
#define GH 4
#define GC 32
#define GHC 128

typedef __attribute__((ext_vector_type(16))) __bf16 v16bf;
typedef __attribute__((ext_vector_type(8)))  float  v8f;

// ---- helpers --------------------------------------------------------------
__device__ __forceinline__ unsigned short f2bf(float f) {
    unsigned u = __float_as_uint(f);
    return (unsigned short)((u + 0x7fffu + ((u >> 16) & 1u)) >> 16);  // RNE
}
// order-preserving float -> uint key (for atomicMax-based segment max)
__device__ __forceinline__ unsigned f2key(float f) {
    unsigned u = __float_as_uint(f);
    return (u >> 31) ? ~u : (u | 0x80000000u);
}
__device__ __forceinline__ float key2f(unsigned k) {
    return (k >> 31) ? __uint_as_float(k & 0x7fffffffu) : __uint_as_float(~k);
}

// ---- generic fills / conversions ------------------------------------------
__global__ void fill_u32_k(unsigned* p, unsigned v, size_t n) {
    size_t i = (size_t)blockIdx.x * blockDim.x + threadIdx.x;
    if (i < n) p[i] = v;
}
__global__ void cvt_bf16_k(const float* in, unsigned short* out, size_t n) {
    size_t i = (size_t)blockIdx.x * blockDim.x + threadIdx.x;
    if (i < n) out[i] = f2bf(in[i]);
}
// out_bf16 = elu(acc + bias[col])  (per-node bias over feature dim F)
__global__ void bias_elu_bf16_k(const float* acc, const float* bias,
                                unsigned short* out, int F, size_t n) {
    size_t i = (size_t)blockIdx.x * blockDim.x + threadIdx.x;
    if (i >= n) return;
    float v = acc[i] + bias[(int)(i % (size_t)F)];
    v = v > 0.f ? v : (__expf(v) - 1.f);
    out[i] = f2bf(v);
}

// ---- WMMA GEMM: C[M,F] = act(A[M,K](bf16) @ W[F,K](bf16)^T + bias) --------
// wave-per-16x16 output tile; K multiple of 32; layouts per CDNA5 ISA 7.12.2
template <int ACT>  // 0 = none, 1 = ELU
__global__ __launch_bounds__(128) void gemm_bf16_wmma(
    const unsigned short* A_, const unsigned short* W_, const float* bias,
    float* Cout, int M, int K, int F) {
    const __bf16* A  = reinterpret_cast<const __bf16*>(A_);
    const __bf16* Wp = reinterpret_cast<const __bf16*>(W_);
    int lane = threadIdx.x & 31;
    int wave = threadIdx.x >> 5;
    int nTilesF = (F + 15) >> 4;
    int mTiles  = (M + 15) >> 4;
    int tile = blockIdx.x * 4 + wave;
    if (tile >= mTiles * nTilesF) return;           // wave-uniform exit
    int mBase = (tile / nTilesF) * 16;
    int nBase = (tile % nTilesF) * 16;
    int half  = lane >> 4;                          // 0: lanes 0-15, 1: 16-31
    int l16   = lane & 15;
    int mRow = mBase + l16;  if (mRow >= M) mRow = M - 1;
    int nRow = nBase + l16;  if (nRow >= F) nRow = F - 1;  // clamp (masked at store)
    const __bf16* arow = A  + (size_t)mRow * K;
    const __bf16* brow = Wp + (size_t)nRow * K;
    const int ga = half * 8;    // A: K-chunk offset per half-wave
    const int gb = half * 8;    // B: K-chunk offset per half-wave
    v8f acc = {};
    for (int k = 0; k < K; k += 32) {
        v16bf a, b;
        // A 16x32: elements 0..7 -> K=k+ga..+7 ; 8..15 -> K=k+16+ga..+7
        reinterpret_cast<uint4*>(&a)[0] = *reinterpret_cast<const uint4*>(arow + k + ga);
        reinterpret_cast<uint4*>(&a)[1] = *reinterpret_cast<const uint4*>(arow + k + 16 + ga);
        // B 32x16: elements 0..7 -> K=k+gb..+7 ; 8..15 -> K=k+16+gb..+7
        reinterpret_cast<uint4*>(&b)[0] = *reinterpret_cast<const uint4*>(brow + k + gb);
        reinterpret_cast<uint4*>(&b)[1] = *reinterpret_cast<const uint4*>(brow + k + 16 + gb);
        acc = __builtin_amdgcn_wmma_f32_16x16x32_bf16(
            false, a, false, b, (short)0, acc, false, false);
    }
    int col = nBase + l16;
    if (col < F) {
        float bv = bias[col];
#pragma unroll
        for (int v = 0; v < 8; ++v) {
            int row = mBase + v + half * 8;
            if (row < M) {
                float r = acc[v] + bv;
                if (ACT == 1) r = r > 0.f ? r : (__expf(r) - 1.f);
                Cout[(size_t)row * F + col] = r;
            }
        }
    }
}

// ---- edge kernels ---------------------------------------------------------
// wave per edge: logit[e,h] = sum_c att[h,c]*leaky(xl[src]+xr[dst]); atomicMax into mkey
__global__ void edge_logit_k(const long long* src, const long long* dst,
                             const float* xl, const float* xr, const float* att,
                             float* logit, unsigned* mkey, int E) {
    int t = blockIdx.x * blockDim.x + threadIdx.x;
    int e = t >> 5, lane = t & 31;
    if (e >= E) return;
    int s = (int)src[e], d = (int)dst[e];
    const float* pl = xl + (size_t)s * GHC;
    const float* pr = xr + (size_t)d * GHC;
    float lg[GH];
#pragma unroll
    for (int h = 0; h < GH; ++h) {
        float v = pl[h * GC + lane] + pr[h * GC + lane];
        v = v > 0.f ? v : 0.2f * v;              // LeakyReLU slope 0.2
        v *= att[h * GC + lane];
#pragma unroll
        for (int off = 16; off; off >>= 1) v += __shfl_xor(v, off, 32);
        lg[h] = v;
    }
    if (lane == 0) {
#pragma unroll
        for (int h = 0; h < GH; ++h) {
            logit[(size_t)e * GH + h] = lg[h];
            atomicMax(&mkey[(size_t)d * GH + h], f2key(lg[h]));
        }
    }
}
// p = exp(logit - max); accumulate segment sums
__global__ void edge_exp_k(const long long* dst, float* logit,
                           const unsigned* mkey, float* ssum, int E) {
    size_t i = (size_t)blockIdx.x * blockDim.x + threadIdx.x;
    if (i >= (size_t)E * GH) return;
    int e = (int)(i >> 2), h = (int)(i & 3);
    int d = (int)dst[e];
    float p = __expf(logit[i] - key2f(mkey[(size_t)d * GH + h]));
    logit[i] = p;
    atomicAdd(&ssum[(size_t)d * GH + h], p);
}
// accum[dst] += xl[src] * alpha   (wave per edge, lane = channel in head)
__global__ void edge_scatter_k(const long long* src, const long long* dst,
                               const float* xl, const float* p, const float* ssum,
                               float* accum, int E) {
    int t = blockIdx.x * blockDim.x + threadIdx.x;
    int e = t >> 5, lane = t & 31;
    if (e >= E) return;
    int s = (int)src[e], d = (int)dst[e];
    const float* xs = xl + (size_t)s * GHC;
    float* ad = accum + (size_t)d * GHC;
#pragma unroll
    for (int h = 0; h < GH; ++h) {
        float alpha = p[(size_t)e * GH + h] /
                      (ssum[(size_t)d * GH + h] + 1e-16f);
        atomicAdd(&ad[h * GC + lane], xs[h * GC + lane] * alpha);
    }
}

// ---------------------------------------------------------------------------
static inline size_t al256(size_t x) { return (x + 255) & ~(size_t)255; }

extern "C" void kernel_launch(void* const* d_in, const int* in_sizes, int n_in,
                              void* d_out, int out_size, void* d_ws, size_t ws_size,
                              hipStream_t stream) {
    (void)in_sizes; (void)n_in; (void)out_size; (void)ws_size;
    const float*     x    = (const float*)d_in[0];
    const long long* ei   = (const long long*)d_in[1];   // int64 [2,E]
    const float* Wproj = (const float*)d_in[2];  const float* bproj = (const float*)d_in[3];
    const float* Wl1 = (const float*)d_in[4];    const float* bl1 = (const float*)d_in[5];
    const float* Wr1 = (const float*)d_in[6];    const float* br1 = (const float*)d_in[7];
    const float* att1 = (const float*)d_in[8];   const float* bias1 = (const float*)d_in[9];
    const float* Wl2 = (const float*)d_in[10];   const float* bl2 = (const float*)d_in[11];
    const float* Wr2 = (const float*)d_in[12];   const float* br2 = (const float*)d_in[13];
    const float* att2 = (const float*)d_in[14];  const float* bias2 = (const float*)d_in[15];
    const float* W1 = (const float*)d_in[16];    const float* b1 = (const float*)d_in[17];
    const float* W2 = (const float*)d_in[18];    const float* b2 = (const float*)d_in[19];
    float* out = (float*)d_out;
    const long long* src = ei;
    const long long* dst = ei + GE;

    // ---- workspace layout (bytes) ----
    uint8_t* ws = (uint8_t*)d_ws;
    size_t o = 0;
    unsigned short* xb16  = (unsigned short*)(ws + o); o += al256((size_t)GN * GDIN * 2);
    unsigned short* wpb   = (unsigned short*)(ws + o); o += al256((size_t)GPROJ * GDIN * 2);
    unsigned short* wl1b  = (unsigned short*)(ws + o); o += al256((size_t)GHC * GPROJ * 2);
    unsigned short* wr1b  = (unsigned short*)(ws + o); o += al256((size_t)GHC * GPROJ * 2);
    unsigned short* wl2b  = (unsigned short*)(ws + o); o += al256((size_t)GHC * GHC * 2);
    unsigned short* wr2b  = (unsigned short*)(ws + o); o += al256((size_t)GHC * GHC * 2);
    unsigned short* w1b   = (unsigned short*)(ws + o); o += al256((size_t)GC * GHC * 2);
    unsigned short* w2b   = (unsigned short*)(ws + o); o += al256((size_t)10 * GC * 2);
    unsigned short* hbf   = (unsigned short*)(ws + o); o += al256((size_t)GN * GPROJ * 2);
    float*          hf32  = (float*)(ws + o);          o += al256((size_t)GN * GPROJ * 4);
    float*          xl    = (float*)(ws + o);          o += al256((size_t)GN * GHC * 4);
    float*          xr    = (float*)(ws + o);          o += al256((size_t)GN * GHC * 4);
    float*          accum = (float*)(ws + o);          o += al256((size_t)GN * GHC * 4);
    float*          logit = (float*)(ws + o);          o += al256((size_t)GE * GH * 4);
    unsigned*       mkey  = (unsigned*)(ws + o);       o += al256((size_t)GN * GH * 4);
    float*          ssum  = (float*)(ws + o);          o += al256((size_t)GN * GH * 4);
    unsigned short* h3b   = (unsigned short*)(ws + o); o += al256((size_t)GN * GC * 2);

    auto CVT = [&](const float* in, unsigned short* outp, size_t n) {
        cvt_bf16_k<<<(unsigned)((n + 255) / 256), 256, 0, stream>>>(in, outp, n);
    };
    auto FILL0 = [&](void* p, size_t nWords) {
        fill_u32_k<<<(unsigned)((nWords + 255) / 256), 256, 0, stream>>>(
            (unsigned*)p, 0u, nWords);
    };
    auto GEMM = [&](int ACT, const unsigned short* A, const unsigned short* Wb,
                    const float* bias, float* Cm, int M, int K, int F) {
        int tiles = ((M + 15) / 16) * ((F + 15) / 16);
        unsigned blocks = (unsigned)((tiles + 3) / 4);
        if (ACT) gemm_bf16_wmma<1><<<blocks, 128, 0, stream>>>(A, Wb, bias, Cm, M, K, F);
        else     gemm_bf16_wmma<0><<<blocks, 128, 0, stream>>>(A, Wb, bias, Cm, M, K, F);
    };
    const unsigned eb  = (GE * 32 + 255) / 256;      // wave-per-edge grids
    const unsigned eb4 = ((size_t)GE * GH + 255) / 256;

    // ---- bf16 conversions of inputs/weights ----
    CVT(x, xb16, (size_t)GN * GDIN);
    CVT(Wproj, wpb, (size_t)GPROJ * GDIN);
    CVT(Wl1, wl1b, (size_t)GHC * GPROJ);  CVT(Wr1, wr1b, (size_t)GHC * GPROJ);
    CVT(Wl2, wl2b, (size_t)GHC * GHC);    CVT(Wr2, wr2b, (size_t)GHC * GHC);
    CVT(W1, w1b, (size_t)GC * GHC);       CVT(W2, w2b, (size_t)10 * GC);

    // ---- projection: h0 = elu(x @ Wproj^T + bproj) ----
    GEMM(1, xb16, wpb, bproj, hf32, GN, GDIN, GPROJ);
    CVT(hf32, hbf, (size_t)GN * GPROJ);

    // ---- GATv2 layer 1 ----
    GEMM(0, hbf, wl1b, bl1, xl, GN, GPROJ, GHC);
    GEMM(0, hbf, wr1b, br1, xr, GN, GPROJ, GHC);
    FILL0(mkey, (size_t)GN * GH); FILL0(ssum, (size_t)GN * GH);
    FILL0(accum, (size_t)GN * GHC);
    edge_logit_k<<<eb, 256, 0, stream>>>(src, dst, xl, xr, att1, logit, mkey, GE);
    edge_exp_k<<<eb4, 256, 0, stream>>>(dst, logit, mkey, ssum, GE);
    edge_scatter_k<<<eb, 256, 0, stream>>>(src, dst, xl, logit, ssum, accum, GE);
    bias_elu_bf16_k<<<(unsigned)(((size_t)GN * GHC + 255) / 256), 256, 0, stream>>>(
        accum, bias1, hbf, GHC, (size_t)GN * GHC);   // h1 (bf16) in hbf

    // ---- GATv2 layer 2 ----
    GEMM(0, hbf, wl2b, bl2, xl, GN, GHC, GHC);
    GEMM(0, hbf, wr2b, br2, xr, GN, GHC, GHC);
    FILL0(mkey, (size_t)GN * GH); FILL0(ssum, (size_t)GN * GH);
    FILL0(accum, (size_t)GN * GHC);
    edge_logit_k<<<eb, 256, 0, stream>>>(src, dst, xl, xr, att2, logit, mkey, GE);
    edge_exp_k<<<eb4, 256, 0, stream>>>(dst, logit, mkey, ssum, GE);
    edge_scatter_k<<<eb, 256, 0, stream>>>(src, dst, xl, logit, ssum, accum, GE);
    bias_elu_bf16_k<<<(unsigned)(((size_t)GN * GHC + 255) / 256), 256, 0, stream>>>(
        accum, bias2, hbf, GHC, (size_t)GN * GHC);   // h2 (bf16) in hbf

    // ---- head MLP ----
    GEMM(1, hbf, w1b, b1, hf32, GN, GHC, GC);        // elu(h2 @ W1^T + b1)
    CVT(hf32, h3b, (size_t)GN * GC);
    GEMM(0, h3b, w2b, b2, out, GN, GC, 10);          // logits -> d_out
}
